// FastNN_9148280340891
// MI455X (gfx1250) — matrix operand order repaired and checked by reading, not compiled, not run on hardware
//
#include <hip/hip_runtime.h>

typedef __attribute__((ext_vector_type(16))) _Float16 v16h;
typedef __attribute__((ext_vector_type(8)))  float    v8f;
typedef __attribute__((ext_vector_type(4)))  float    f4;

#define EPSV 1e-5f

// ---- WMMA fragment loaders (layouts per CDNA5 ISA 7.12.2) ----

// A-matrix 16x32 f16: lane L -> M=L%16; lanes 0-15: K {0..7} in v0-3, {16..23} in v4-7;
// lanes 16-31: K {8..15}, {24..31}.  LDS activation layout: [pix][32] f16 (K contiguous).
// `base` is a per-lane pre-offset pointer (already includes pix*32 + hi*8); `off` is a
// compile-time-constant half offset so ds_load_b128 gets immediate offsets.
__device__ __forceinline__ v16h ldsA_frag(const _Float16* base, int off, bool valid) {
    union { v16h h; f4 q[2]; } u;
    if (valid) {
        const f4* p = (const f4*)(base + off);
        u.q[0] = p[0];   // K = 8*hi .. 8*hi+7
        u.q[1] = p[2];   // K = 16+8*hi .. 16+8*hi+7
    } else {
        u.h = v16h{};
    }
    return u.h;
}

// B-matrix 32x16 f16: lane L -> N=L%16; lanes 0-15: K {0..7},{16..23}; lanes 16-31: K {8..15},{24..31}.
// LDS weight layout: [N][32] f16 (K contiguous) -> two 16B reads per lane.
__device__ __forceinline__ v16h ldsB_frag(const _Float16* wbase, int off) {
    union { v16h h; f4 q[2]; } u;
    const f4* p = (const f4*)(wbase + off);
    u.q[0] = p[0];
    u.q[1] = p[2];
    return u.h;
}

// ---- Fused sparse conv block: channel path + WMMA bottleneck + select + maxpool2 ----
// One workgroup per 16x16 spatial block. 256 threads = 8 wave32.
template <int CIN, int C0, int C1, int HH, int WW, bool MPOOL>
__global__ __launch_bounds__(256) void sparse_stage(
    const float* __restrict__ x,    const float* __restrict__ mask,
    const float* __restrict__ cw,   const float* __restrict__ cb,   const float* __restrict__ bn0,
    const float* __restrict__ d1w,  const float* __restrict__ d1b,  const float* __restrict__ bn1,
    const float* __restrict__ d2w,  const float* __restrict__ d2b,  const float* __restrict__ bn2,
    const float* __restrict__ d3w,  const float* __restrict__ d3b,  const float* __restrict__ bn3,
    float* __restrict__ out)
{
    constexpr int NT  = C1 / 16;        // N-tiles for d1/d2 outputs
    constexpr int NBX = WW / 16;
    constexpr int NBY = HH / 16;
    // folded (bias, scale, shift) table offsets
    constexpr int T0 = 0;
    constexpr int T1 = C0;
    constexpr int T2 = C0 + C1;
    constexpr int T3 = C0 + 2 * C1;
    constexpr int TN = 2 * C0 + 2 * C1;

    __shared__ _Float16 actA[256 * 32];       // ping: xc(padded) -> g2
    __shared__ _Float16 actB[256 * 32];       // pong: g1 -> g3
    __shared__ _Float16 xc16[256 * C0];       // channel-path result (for inactive blocks)
    __shared__ _Float16 w1L[C1 * 32];         // d1 weights [N][K]
    __shared__ _Float16 w2L[9 * C1 * 32];     // d2 weights [tap][N][K]
    __shared__ _Float16 w3L[16 * 32];         // d3 weights [N(pad16)][K]
    __shared__ f4       bnt[TN];              // per-channel folded (bias, scale, shift, 0)
    __shared__ float    red[256];
    __shared__ int      s_active;

    const int tid  = threadIdx.x;
    const int wid  = tid >> 5;
    const int lane = tid & 31;
    const int hi   = lane >> 4;
    const int lm   = lane & 15;

    const int bid = blockIdx.x;
    const int n   = bid / (NBY * NBX);
    const int rem = bid % (NBY * NBX);
    const int by  = rem / NBX;
    const int bx  = rem % NBX;

    // ---- Phase 0: zero pong buffer, stage weights, fold BN params, mask value ----
    for (int i = tid; i < 256 * 16; i += 256) ((unsigned*)actB)[i] = 0u;
    for (int i = tid; i < C1 * 32; i += 256) {
        int nn = i >> 5, k = i & 31;
        w1L[i] = (k < C0) ? (_Float16)d1w[nn * C0 + k] : (_Float16)0.f;
    }
    for (int i = tid; i < 9 * C1 * 32; i += 256) {
        int t = i / (C1 * 32), r = i % (C1 * 32);
        int nn = r >> 5, k = r & 31;
        w2L[i] = (k < C1) ? (_Float16)d2w[(nn * C1 + k) * 9 + t] : (_Float16)0.f;
    }
    for (int i = tid; i < 16 * 32; i += 256) {
        int nn = i >> 5, k = i & 31;
        w3L[i] = (nn < C0 && k < C1) ? (_Float16)d3w[nn * C1 + k] : (_Float16)0.f;
    }
    if (tid < TN) {
        const float* bn; const float* bs; int ch; int C;
        if      (tid < T1) { bn = bn0; bs = cb;  ch = tid;      C = C0; }
        else if (tid < T2) { bn = bn1; bs = d1b; ch = tid - T1; C = C1; }
        else if (tid < T3) { bn = bn2; bs = d2b; ch = tid - T2; C = C1; }
        else               { bn = bn3; bs = d3b; ch = tid - T3; C = C0; }
        float g = bn[ch], bt = bn[C + ch], m = bn[2 * C + ch], vv = bn[3 * C + ch];
        float sc = g * rsqrtf(vv + EPSV);
        f4 e; e.x = bs[ch]; e.y = sc; e.z = bt - sc * m; e.w = 0.f;
        bnt[tid] = e;
    }

    {   // mask value for this thread's pixel (stage2 pools mask 2x2 on the fly)
        int py = tid >> 4, px = tid & 15;
        int gy = by * 16 + py, gx = bx * 16 + px;
        float mv;
        if (!MPOOL) {
            mv = mask[(size_t)n * HH * WW + (size_t)gy * WW + gx];
        } else {
            const float* mp = mask + (size_t)n * (2 * HH) * (2 * WW);
            int r0 = 2 * gy, c0 = 2 * gx;
            float a = mp[(size_t)r0 * (2 * WW) + c0];
            float b = mp[(size_t)r0 * (2 * WW) + c0 + 1];
            float c = mp[(size_t)(r0 + 1) * (2 * WW) + c0];
            float d = mp[(size_t)(r0 + 1) * (2 * WW) + c0 + 1];
            mv = fmaxf(fmaxf(a, b), fmaxf(c, d));
        }
        red[tid] = mv;
    }
    __syncthreads();
    for (int s = 128; s > 0; s >>= 1) {
        if (tid < s) red[tid] += red[tid + s];
        __syncthreads();
    }
    if (tid == 0) s_active = (red[0] > 128.0f) ? 1 : 0;  // mean > 0.5

    // ---- Phase 1: channel path (1x1 conv CIN->C0, relu, bn0) -> actA (K-padded) + xc16 ----
    {
        int py = tid >> 4, px = tid & 15;
        int gy = by * 16 + py, gx = bx * 16 + px;
        float xv[CIN];
#pragma unroll
        for (int ci = 0; ci < CIN; ci++)
            xv[ci] = x[(((size_t)n * CIN + ci) * HH + gy) * WW + gx];
#pragma unroll
        for (int co = 0; co < 32; co++) {
            _Float16 hv = (_Float16)0.f;
            if (co < C0) {
                f4 e = bnt[T0 + co];
                float a = e.x;
#pragma unroll
                for (int ci = 0; ci < CIN; ci++) a += cw[co * CIN + ci] * xv[ci];
                float r = e.y * fmaxf(a, 0.f) + e.z;
                hv = (_Float16)r;
                xc16[tid * C0 + co] = hv;
            }
            actA[tid * 32 + co] = hv;
        }
    }
    __syncthreads();

    // ---- Phase 2: d1 = bn1(relu(1x1 C0->C1)) via WMMA, actA -> actB ----
#pragma unroll
    for (int yy = 0; yy < 2; yy++) {
        const int y = wid * 2 + yy;
        const _Float16* aBase = actA + (y * 16 + lm) * 32 + hi * 8;
        v16h A = ldsA_frag(aBase, 0, true);
#pragma unroll
        for (int nt = 0; nt < NT; nt++) {
            const int ch = nt * 16 + lm;
            v16h B = ldsB_frag(w1L + ch * 32 + hi * 8, 0);
            v8f  c = {};
            c = __builtin_amdgcn_wmma_f32_16x16x32_f16(false, A, false, B, (short)0, c, false, false);
            f4 e = bnt[T1 + ch];
            _Float16* p = actB + (y * 16 + 8 * hi) * 32 + ch;   // + v*32 per element
#pragma unroll
            for (int v = 0; v < 8; v++)
                p[v * 32] = (_Float16)(e.y * fmaxf(c[v] + e.x, 0.f) + e.z);
        }
    }
    __syncthreads();

    // ---- Phase 3: d2 = bn2(relu(3x3 C1->C1, zero-padded within block)), actB -> actA ----
#pragma unroll
    for (int yy = 0; yy < 2; yy++) {
        const int y = wid * 2 + yy;
        // base biased by (-1,-1) so tap offsets (dy1*16+dx1)*32 are non-negative constants
        const _Float16* aBase = actB + ((y - 1) * 16 + (lm - 1)) * 32 + hi * 8;
        const _Float16* wBase = w2L + lm * 32 + hi * 8;
        v8f acc[NT];
#pragma unroll
        for (int nt = 0; nt < NT; nt++) { v8f z = {}; acc[nt] = z; }
#pragma unroll
        for (int t = 0; t < 9; t++) {
            const int dy1 = t / 3, dx1 = t % 3;                 // 0..2
            const int yr = y + dy1 - 1, xr = lm + dx1 - 1;
            const bool ok = ((unsigned)yr < 16u) && ((unsigned)xr < 16u);
            v16h A = ldsA_frag(aBase, (dy1 * 16 + dx1) * 32, ok);
#pragma unroll
            for (int nt = 0; nt < NT; nt++) {
                v16h B = ldsB_frag(wBase, (t * C1 + nt * 16) * 32);
                acc[nt] = __builtin_amdgcn_wmma_f32_16x16x32_f16(false, A, false, B, (short)0, acc[nt], false, false);
            }
        }
#pragma unroll
        for (int nt = 0; nt < NT; nt++) {
            const int ch = nt * 16 + lm;
            f4 e = bnt[T2 + ch];
            _Float16* p = actA + (y * 16 + 8 * hi) * 32 + ch;
#pragma unroll
            for (int v = 0; v < 8; v++)
                p[v * 32] = (_Float16)(e.y * fmaxf(acc[nt][v] + e.x, 0.f) + e.z);
        }
    }
    __syncthreads();   // all d2 reads of actB done before d3 overwrites it

    // ---- Phase 4: d3 = bn3(relu(1x1 C1->C0)) via WMMA, actA -> actB ----
#pragma unroll
    for (int yy = 0; yy < 2; yy++) {
        const int y = wid * 2 + yy;
        const _Float16* aBase = actA + (y * 16 + lm) * 32 + hi * 8;
        v16h A = ldsA_frag(aBase, 0, true);
        v16h B = ldsB_frag(w3L + lm * 32 + hi * 8, 0);
        v8f  c = {};
        c = __builtin_amdgcn_wmma_f32_16x16x32_f16(false, A, false, B, (short)0, c, false, false);
        const int ch = lm;
        if (ch < C0) {
            f4 e = bnt[T3 + ch];
            _Float16* p = actB + (y * 16 + 8 * hi) * 32 + ch;
#pragma unroll
            for (int v = 0; v < 8; v++)
                p[v * 32] = (_Float16)(e.y * fmaxf(c[v] + e.x, 0.f) + e.z);
        }
    }
    __syncthreads();

    // ---- Phase 5: select (active ? g3 : xc) + fused maxpool 2x2 -> global ----
    constexpr int OH = HH / 2, OW = WW / 2;
    const bool act = (s_active != 0);
    for (int i = tid; i < C0 * 64; i += 256) {
        int ch = i >> 6;
        int rr = i & 63;
        int py = rr >> 3, px = rr & 7;
        int p00 = (2 * py) * 16 + 2 * px;
        float a, b, c, d;
        if (act) {
            const _Float16* p = actB + p00 * 32 + ch;
            a = (float)p[0];  b = (float)p[32];  c = (float)p[16 * 32];  d = (float)p[17 * 32];
        } else {
            const _Float16* p = xc16 + p00 * C0 + ch;
            a = (float)p[0];  b = (float)p[C0];  c = (float)p[16 * C0];  d = (float)p[17 * C0];
        }
        float mx = fmaxf(fmaxf(a, b), fmaxf(c, d));
        int oy = by * 8 + py, ox = bx * 8 + px;
        out[(((size_t)n * C0 + ch) * OH + oy) * OW + ox] = mx;
    }
}

// ---- FC: logits[n][j] = dot(flat[n], fc_w[j]) + fc_b[j]; one WG per (n,j) ----
__global__ __launch_bounds__(256) void fc_kernel(const float* __restrict__ feat,
                                                 const float* __restrict__ w,
                                                 const float* __restrict__ b,
                                                 float* __restrict__ logits,
                                                 int nfeat)
{
    __shared__ float red[256];
    const int tid = threadIdx.x;
    const int n = blockIdx.x / 10, j = blockIdx.x % 10;
    const float* f  = feat + (size_t)n * nfeat;
    const float* wr = w + (size_t)j * nfeat;
    float s = 0.f;
    for (int i = tid; i < nfeat; i += 256) s += f[i] * wr[i];
    red[tid] = s;
    __syncthreads();
    for (int st = 128; st > 0; st >>= 1) {
        if (tid < st) red[tid] += red[tid + st];
        __syncthreads();
    }
    if (tid == 0) logits[n * 10 + j] = red[0] + b[j];
}

__global__ void softmax_kernel(const float* __restrict__ logits, float* __restrict__ out)
{
    int n = threadIdx.x;
    if (n < 16) {
        float l[10], mx = -1e30f;
#pragma unroll
        for (int j = 0; j < 10; j++) { l[j] = logits[n * 10 + j]; mx = fmaxf(mx, l[j]); }
        float s = 0.f;
#pragma unroll
        for (int j = 0; j < 10; j++) { l[j] = expf(l[j] - mx); s += l[j]; }
        float inv = 1.f / s;
#pragma unroll
        for (int j = 0; j < 10; j++) out[n * 10 + j] = l[j] * inv;
    }
}

extern "C" void kernel_launch(void* const* d_in, const int* in_sizes, int n_in,
                              void* d_out, int out_size, void* d_ws, size_t ws_size,
                              hipStream_t stream)
{
    (void)in_sizes; (void)n_in; (void)out_size; (void)ws_size;
    const float* x      = (const float*)d_in[0];
    const float* mask   = (const float*)d_in[1];
    const float* s1_cw  = (const float*)d_in[2];
    const float* s1_cb  = (const float*)d_in[3];
    const float* s1_bn0 = (const float*)d_in[4];
    const float* s1_d1w = (const float*)d_in[5];
    const float* s1_d1b = (const float*)d_in[6];
    const float* s1_bn1 = (const float*)d_in[7];
    const float* s1_d2w = (const float*)d_in[8];
    const float* s1_d2b = (const float*)d_in[9];
    const float* s1_bn2 = (const float*)d_in[10];
    const float* s1_d3w = (const float*)d_in[11];
    const float* s1_d3b = (const float*)d_in[12];
    const float* s1_bn3 = (const float*)d_in[13];
    const float* s2_cw  = (const float*)d_in[14];
    const float* s2_cb  = (const float*)d_in[15];
    const float* s2_bn0 = (const float*)d_in[16];
    const float* s2_d1w = (const float*)d_in[17];
    const float* s2_d1b = (const float*)d_in[18];
    const float* s2_bn1 = (const float*)d_in[19];
    const float* s2_d2w = (const float*)d_in[20];
    const float* s2_d2b = (const float*)d_in[21];
    const float* s2_bn2 = (const float*)d_in[22];
    const float* s2_d3w = (const float*)d_in[23];
    const float* s2_d3b = (const float*)d_in[24];
    const float* s2_bn3 = (const float*)d_in[25];
    const float* fc_w   = (const float*)d_in[26];
    const float* fc_b   = (const float*)d_in[27];

    float* ws     = (float*)d_ws;
    float* x1     = ws;                                 // [16,16,256,256] = 16777216 f32
    float* x2     = ws + 16777216;                      // [16, 8,128,128] =  2097152 f32
    float* logits = x2 + 2097152;                       // [16,10]

    // Stage 1: 16 * 32 * 32 blocks of 16x16 over 512x512, CIN=3, C0=16, C1=32
    sparse_stage<3, 16, 32, 512, 512, false><<<16 * 32 * 32, 256, 0, stream>>>(
        x, mask, s1_cw, s1_cb, s1_bn0, s1_d1w, s1_d1b, s1_bn1,
        s1_d2w, s1_d2b, s1_bn2, s1_d3w, s1_d3b, s1_bn3, x1);

    // Stage 2: 16 * 16 * 16 blocks over 256x256, CIN=16, C0=8, C1=16, mask pooled on the fly
    sparse_stage<16, 8, 16, 256, 256, true><<<16 * 16 * 16, 256, 0, stream>>>(
        x1, mask, s2_cw, s2_cb, s2_bn0, s2_d1w, s2_d1b, s2_bn1,
        s2_d2w, s2_d2b, s2_bn2, s2_d3w, s2_d3b, s2_bn3, x2);

    // FC (16 rows x 10 outputs, K = 131072) + softmax
    fc_kernel<<<160, 256, 0, stream>>>(x2, fc_w, fc_b, logits, 128 * 128 * 8);
    softmax_kernel<<<1, 32, 0, stream>>>(logits, (float*)d_out);
}
